// TemporalGNN_85169201479940
// MI455X (gfx1250) — compile-verified, matrix operand order checked by reading, not run on hardware
//
#include <hip/hip_runtime.h>
#include <math.h>

#define NB 4
#define NT 12
#define NN 10000
#define NF 32
#define NE 160000
#define RN (NB * NN)            // 40000 flattened (b,n) rows
#define ROW_TILES (RN / 16)     // 2500 (exact)

typedef __attribute__((ext_vector_type(16))) _Float16 v16h;
typedef __attribute__((ext_vector_type(8)))  float    v8f;

union H16 { v16h v; _Float16 h[16]; };
union F8  { v8f  v; float    f[8];  };

__device__ __forceinline__ v8f wmma_f16(v16h a, v16h b, v8f c) {
  // D = A(16x32 f16) * B(32x16 f16) + C(16x16 f32)
  return __builtin_amdgcn_wmma_f32_16x16x32_f16(
      /*neg_a=*/false, a, /*neg_b=*/false, b,
      /*c_mod=*/(short)0, c, /*reuse_a=*/false, /*reuse_b=*/false);
}

// ---------------------------------------------------------------- prep kernels

__global__ void fill_f32(float* p, float v, int count) {
  int gid = blockIdx.x * blockDim.x + threadIdx.x;
  if (gid < count) p[gid] = v;
}

__global__ void deg_scatter(const int* coli, const float* ew, float* deg) {
  int e = blockIdx.x * blockDim.x + threadIdx.x;
  if (e < NE) atomicAdd(&deg[coli[e]], ew[e]);
}

__global__ void dinv_kernel(const float* deg, float* dinv, float* selfn) {
  int n = blockIdx.x * blockDim.x + threadIdx.x;
  if (n < NN) {
    float d  = deg[n];
    float di = (d > 0.f) ? rsqrtf(fmaxf(d, 1e-12f)) : 0.f;
    dinv[n]  = di;
    selfn[n] = di * di;   // self-loop weight 1.0 * di * di
  }
}

__global__ void norm_kernel(const int* rowi, const int* coli, const float* ew,
                            const float* dinv, float* normw) {
  int e = blockIdx.x * blockDim.x + threadIdx.x;
  if (e < NE) normw[e] = dinv[rowi[e]] * ew[e] * dinv[coli[e]];
}

__global__ void softmax12(const float* att, float* probs) {
  if (threadIdx.x == 0 && blockIdx.x == 0) {
    float m = -1e30f;
    for (int i = 0; i < NT; ++i) m = fmaxf(m, att[i]);
    float ex[NT]; float s = 0.f;
    for (int i = 0; i < NT; ++i) { ex[i] = expf(att[i] - m); s += ex[i]; }
    float inv = 1.f / s;
    for (int i = 0; i < NT; ++i) probs[i] = ex[i] * inv;
  }
}

// Pack all weight matrices into per-lane WMMA B-layout:
// tile j, lane L -> 16 contiguous f16: col n = (L&15)+16*colTile,
// K values {k0..k0+7, k0+16..k0+23} with k0 = (L>>4)*8.
// tiles 0-5:  [Wz|Wr|Wh]  (g=tile/2, colTile=tile&1, rowOff=0)
// tiles 6-17: gate linears (g, kchunk, colTile), rowOff = kchunk*32
// tiles 18-19: lin_w
__global__ void pack_weights(const float* Wz, const float* Wr, const float* Wh,
                             const float* lzw, const float* lrw, const float* lhw,
                             const float* linw, _Float16* packed) {
  int tile = blockIdx.x;
  int lane = threadIdx.x;
  const float* src; int rowOff = 0; int ct;
  if (tile < 6) {
    int g = tile >> 1; ct = tile & 1;
    src = (g == 0) ? Wz : (g == 1) ? Wr : Wh;
  } else if (tile < 18) {
    int idx = tile - 6; int g = idx >> 2; int rem = idx & 3;
    int kc = rem >> 1;  ct = rem & 1;
    src = (g == 0) ? lzw : (g == 1) ? lrw : lhw;
    rowOff = kc * 32;
  } else {
    ct = tile - 18; src = linw;
  }
  int n  = (lane & 15) + 16 * ct;
  int k0 = (lane >> 4) * 8;
  _Float16* dst = packed + (size_t)tile * 512 + (size_t)lane * 16;
  for (int i = 0; i < 8; ++i) {
    dst[i]     = (_Float16)src[(rowOff + k0 + i) * 32 + n];
    dst[8 + i] = (_Float16)src[(rowOff + k0 + 16 + i) * 32 + n];
  }
}

// ---------------------------------------------------------- per-timestep kernels

// sup[r, 0:96] = x_t[r, :] @ [Wz|Wr|Wh]   (one wave per 16-row tile, 6 WMMAs)
__global__ __launch_bounds__(32)
void sup_mm(const float* x, const _Float16* packed, float* sup, int t) {
  int lane = threadIdx.x;
  int rowBase = blockIdx.x * 16;
  int lm = lane & 15, half = lane >> 4, k0 = half * 8;
  int r = rowBase + lm;
  int b = r / NN, n = r % NN;
  const float* xrow = x + (((size_t)b * NT + t) * NN + n) * 32;
  if (t + 1 < NT) __builtin_prefetch(xrow + (size_t)NN * 32, 0, 0);
  H16 a;
  for (int i = 0; i < 8; ++i) {
    a.h[i]     = (_Float16)xrow[k0 + i];
    a.h[8 + i] = (_Float16)xrow[k0 + 16 + i];
  }
  for (int tile = 0; tile < 6; ++tile) {
    H16 bm;
    bm.v = *(const v16h*)(packed + (size_t)tile * 512 + (size_t)lane * 16);
    F8 c; for (int i = 0; i < 8; ++i) c.f[i] = 0.f;
    c.v = wmma_f16(a.v, bm.v, c.v);
    int g = tile >> 1, ct = tile & 1;
    int col = g * 32 + ct * 16 + lm;
    for (int jv = 0; jv < 8; ++jv)
      sup[(size_t)(rowBase + jv + 8 * half) * 96 + col] = c.f[jv];
  }
}

// agg = sup * selfnorm + bias (self-loop contribution + GCN bias)
__global__ void agg_init(const float* sup, const float* selfn,
                         const float* bz, const float* br, const float* bh,
                         float* agg) {
  long long gid = (long long)blockIdx.x * blockDim.x + threadIdx.x;
  if (gid >= (long long)RN * 96) return;
  int r = (int)(gid / 96), c = (int)(gid % 96);
  int n = r % NN;
  float bias = (c < 32) ? bz[c] : (c < 64) ? br[c - 32] : bh[c - 64];
  agg[gid] = sup[gid] * selfn[n] + bias;
}

// agg[b, col[e], c] += sup[b, row[e], c] * norm[e]   (f32 atomics, L2-resident)
__global__ void edge_scatter(const int* rowi, const int* coli, const float* normw,
                             const float* sup, float* agg) {
  long long gid = (long long)blockIdx.x * blockDim.x + threadIdx.x;
  if (gid >= (long long)NE * NB * 96) return;
  int e   = (int)(gid / 384);
  int rem = (int)(gid % 384);
  int b = rem / 96, c = rem % 96;
  float v = sup[((size_t)b * NN + rowi[e]) * 96 + c] * normw[e];
  atomicAdd(&agg[((size_t)b * NN + coli[e]) * 96 + c], v);
}

// Fused GRU gate update: Z, R, Ht via WMMA (K=64 as two K=32 chains),
// sigmoid/tanh, H := Z*H + (1-Z)*Ht, Hacc += p_t * H.
__global__ __launch_bounds__(32)
void gate_update(const float* agg, float* Hst, float* Hacc, const _Float16* packed,
                 const float* lzb, const float* lrb, const float* lhb,
                 const float* probs, int t) {
  __shared__ float Rbuf[16][33];   // +1 pad: dodge LDS bank conflicts
  int lane = threadIdx.x;
  int rowBase = blockIdx.x * 16;
  int lm = lane & 15, half = lane >> 4, k0 = half * 8;
  int r = rowBase + lm;
  float p = probs[t];

  float hrow[16];
  H16 aH, aZ, aR, aHin;
  const float* hp = Hst + (size_t)r * 32;
  const float* ap = agg + (size_t)r * 96;
  for (int i = 0; i < 8; ++i) {
    int ka = k0 + i, kb = k0 + 16 + i;
    hrow[i]     = hp[ka];
    hrow[8 + i] = hp[kb];
    aH.h[i]   = (_Float16)hrow[i];      aH.h[8 + i]   = (_Float16)hrow[8 + i];
    aZ.h[i]   = (_Float16)ap[ka];       aZ.h[8 + i]   = (_Float16)ap[kb];
    aR.h[i]   = (_Float16)ap[32 + ka];  aR.h[8 + i]   = (_Float16)ap[32 + kb];
    aHin.h[i] = (_Float16)ap[64 + ka];  aHin.h[8 + i] = (_Float16)ap[64 + kb];
  }
  auto Bt = [&](int g, int kc, int ct) -> v16h {
    int tile = 6 + g * 4 + kc * 2 + ct;
    return *(const v16h*)(packed + (size_t)tile * 512 + (size_t)lane * 16);
  };

  float zval[2][8], htval[2][8];
  for (int ct = 0; ct < 2; ++ct) {
    F8 cz; for (int i = 0; i < 8; ++i) cz.f[i] = 0.f;
    cz.v = wmma_f16(aZ.v, Bt(0, 0, ct), cz.v);
    cz.v = wmma_f16(aH.v, Bt(0, 1, ct), cz.v);
    F8 cr; for (int i = 0; i < 8; ++i) cr.f[i] = 0.f;
    cr.v = wmma_f16(aR.v, Bt(1, 0, ct), cr.v);
    cr.v = wmma_f16(aH.v, Bt(1, 1, ct), cr.v);
    int col = lm + 16 * ct;
    float bZ = lzb[col], bR = lrb[col];
    for (int jv = 0; jv < 8; ++jv) {
      float z  = 1.f / (1.f + expf(-(cz.f[jv] + bZ)));
      float rr = 1.f / (1.f + expf(-(cr.f[jv] + bR)));
      zval[ct][jv] = z;
      Rbuf[jv + 8 * half][col] = rr;    // D-layout -> plain [row][col] in LDS
    }
  }
  __syncthreads();

  H16 aHR;                              // A-layout of H * sigmoid(R)
  for (int i = 0; i < 8; ++i) {
    int ka = k0 + i, kb = k0 + 16 + i;
    aHR.h[i]     = (_Float16)(hrow[i]     * Rbuf[lm][ka]);
    aHR.h[8 + i] = (_Float16)(hrow[8 + i] * Rbuf[lm][kb]);
  }
  for (int ct = 0; ct < 2; ++ct) {
    F8 ch; for (int i = 0; i < 8; ++i) ch.f[i] = 0.f;
    ch.v = wmma_f16(aHin.v, Bt(2, 0, ct), ch.v);
    ch.v = wmma_f16(aHR.v,  Bt(2, 1, ct), ch.v);
    int col = lm + 16 * ct;
    float bH = lhb[col];
    for (int jv = 0; jv < 8; ++jv) htval[ct][jv] = tanhf(ch.f[jv] + bH);
  }
  for (int ct = 0; ct < 2; ++ct) {
    int col = lm + 16 * ct;
    for (int jv = 0; jv < 8; ++jv) {
      int m = jv + 8 * half;
      size_t idx = (size_t)(rowBase + m) * 32 + col;
      float hold = Hst[idx];            // pre-update H at D-layout position
      float z = zval[ct][jv];
      float hn = z * hold + (1.f - z) * htval[ct][jv];
      Hst[idx] = hn;
      Hacc[idx] += p * hn;
    }
  }
}

// out = relu(Hacc) @ lin_w + lin_b
__global__ __launch_bounds__(32)
void final_mm(const float* Hacc, const _Float16* packed, const float* linb,
              float* out) {
  int lane = threadIdx.x;
  int rowBase = blockIdx.x * 16;
  int lm = lane & 15, half = lane >> 4, k0 = half * 8;
  int r = rowBase + lm;
  const float* hp = Hacc + (size_t)r * 32;
  H16 a;
  for (int i = 0; i < 8; ++i) {
    a.h[i]     = (_Float16)fmaxf(hp[k0 + i], 0.f);
    a.h[8 + i] = (_Float16)fmaxf(hp[k0 + 16 + i], 0.f);
  }
  for (int ct = 0; ct < 2; ++ct) {
    F8 c; for (int i = 0; i < 8; ++i) c.f[i] = 0.f;
    c.v = wmma_f16(a.v,
        *(const v16h*)(packed + (size_t)(18 + ct) * 512 + (size_t)lane * 16),
        c.v);
    int col = lm + 16 * ct;
    float bb = linb[col];
    for (int jv = 0; jv < 8; ++jv)
      out[(size_t)(rowBase + jv + 8 * half) * 32 + col] = c.f[jv] + bb;
  }
}

// ------------------------------------------------------------------- launcher

extern "C" void kernel_launch(void* const* d_in, const int* in_sizes, int n_in,
                              void* d_out, int out_size, void* d_ws, size_t ws_size,
                              hipStream_t stream) {
  const float* x    = (const float*)d_in[0];
  const int*   ei   = (const int*)d_in[1];
  const float* ew   = (const float*)d_in[2];
  const float* Wz   = (const float*)d_in[3];
  const float* bz   = (const float*)d_in[4];
  const float* lzw  = (const float*)d_in[5];
  const float* lzb  = (const float*)d_in[6];
  const float* Wr   = (const float*)d_in[7];
  const float* br   = (const float*)d_in[8];
  const float* lrw  = (const float*)d_in[9];
  const float* lrb  = (const float*)d_in[10];
  const float* Wh   = (const float*)d_in[11];
  const float* bh   = (const float*)d_in[12];
  const float* lhw  = (const float*)d_in[13];
  const float* lhb  = (const float*)d_in[14];
  const float* att  = (const float*)d_in[15];
  const float* linw = (const float*)d_in[16];
  const float* linb = (const float*)d_in[17];
  float* out = (float*)d_out;
  const int* rowi = ei;        // edge_index[0]
  const int* coli = ei + NE;   // edge_index[1]

  // workspace layout (packed base lands 32B-aligned: (3*NN+NE+16)*4 % 32 == 0)
  float* deg      = (float*)d_ws;                 // NN
  float* dinv     = deg + NN;                     // NN
  float* selfn    = dinv + NN;                    // NN
  float* normw    = selfn + NN;                   // NE
  float* probs    = normw + NE;                   // 16
  _Float16* packed = (_Float16*)(probs + 16);     // 20 tiles * 512 f16
  float* sup      = (float*)(packed + 20 * 512);  // RN*96
  float* agg      = sup + (size_t)RN * 96;        // RN*96
  float* Hst      = agg + (size_t)RN * 96;        // RN*32
  float* Hacc     = Hst + (size_t)RN * 32;        // RN*32

  fill_f32<<<(RN * 32 + 255) / 256, 256, 0, stream>>>(Hst, 0.f, RN * 32);
  fill_f32<<<(RN * 32 + 255) / 256, 256, 0, stream>>>(Hacc, 0.f, RN * 32);
  fill_f32<<<(NN + 255) / 256, 256, 0, stream>>>(deg, 1.0f, NN);  // self-loop ew
  deg_scatter<<<(NE + 255) / 256, 256, 0, stream>>>(coli, ew, deg);
  dinv_kernel<<<(NN + 255) / 256, 256, 0, stream>>>(deg, dinv, selfn);
  norm_kernel<<<(NE + 255) / 256, 256, 0, stream>>>(rowi, coli, ew, dinv, normw);
  softmax12<<<1, 32, 0, stream>>>(att, probs);
  pack_weights<<<20, 32, 0, stream>>>(Wz, Wr, Wh, lzw, lrw, lhw, linw, packed);

  const long long aggN  = (long long)RN * 96;
  const long long scatN = (long long)NE * NB * 96;
  for (int t = 0; t < NT; ++t) {
    sup_mm<<<ROW_TILES, 32, 0, stream>>>(x, packed, sup, t);
    agg_init<<<(int)((aggN + 255) / 256), 256, 0, stream>>>(sup, selfn, bz, br, bh, agg);
    edge_scatter<<<(int)((scatN + 255) / 256), 256, 0, stream>>>(rowi, coli, normw, sup, agg);
    gate_update<<<ROW_TILES, 32, 0, stream>>>(agg, Hst, Hacc, packed, lzb, lrb, lhb, probs, t);
  }
  final_mm<<<ROW_TILES, 32, 0, stream>>>(Hacc, packed, linb, out);
}
